// MixtralBlock_40106404610207
// MI455X (gfx1250) — matrix operand order repaired and checked by reading, not compiled, not run on hardware
//
#include <hip/hip_runtime.h>

typedef unsigned short u16;
typedef __attribute__((ext_vector_type(16))) __bf16 v16bf;
typedef __attribute__((ext_vector_type(8)))  float  v8f;
typedef __attribute__((ext_vector_type(4)))  int    v4i;
typedef __attribute__((ext_vector_type(4)))  unsigned uv4;
typedef __attribute__((ext_vector_type(8)))  int    iv8;
typedef __attribute__((ext_vector_type(4)))  int    iv4;

namespace {
constexpr int cB = 2, cS = 2048, cD = 2048;
constexpr int cH = 16, cKV = 4, cHD = 128;
constexpr int cE = 8, cF = 4096;
constexpr int cT = cB * cS;   // 4096 tokens

struct alignas(16) P16 { unsigned x, y, z, w; };
struct alignas(16) F4  { float x, y, z, w; };
union Frag { v16bf v; P16 q[2]; };
}

// ---------------- CDNA5 async global->LDS path (guarded, sync fallback) ----
#if defined(__has_builtin)
#if __has_builtin(__builtin_amdgcn_global_load_async_to_lds_b128) && \
    __has_builtin(__builtin_amdgcn_s_wait_asynccnt)
#define USE_ASYNC_LDS 1
#endif
#if __has_builtin(__builtin_amdgcn_tensor_load_to_lds) && \
    __has_builtin(__builtin_amdgcn_s_wait_tensorcnt)
#define USE_TDM 1
#endif
#endif
#ifndef USE_ASYNC_LDS
#define USE_ASYNC_LDS 0
#endif
#ifndef USE_TDM
#define USE_TDM 0
#endif

typedef __attribute__((address_space(1))) v4i gv4i;   // global int4
typedef __attribute__((address_space(3))) v4i lv4i;   // LDS int4

__device__ __forceinline__ void copy16_to_lds(u16* l, const u16* g) {
#if USE_ASYNC_LDS
  __builtin_amdgcn_global_load_async_to_lds_b128((gv4i*)g, (lv4i*)l, 0, 0);
#else
  *(P16*)l = *(const P16*)g;
#endif
}
__device__ __forceinline__ void async_wait() {
#if USE_ASYNC_LDS
  __builtin_amdgcn_s_wait_asynccnt(0);
#endif
}

// TDM: 2D tile of bf16, tile_dim 32 x rows(<=128), row stride lda elements,
// LDS padding: every 16 DWORDs (32 bf16) insert 4 DWORDs (8 bf16) -> stride 40.
// OOB rows (y >= dim1) are zero-filled by the TDM.
__device__ __forceinline__ void tdm_load_A(u16* lds, const u16* gsrc,
                                           int rows, int lda) {
#if USE_TDM
  unsigned lds_addr = (unsigned)(size_t)(lv4i*)lds;
  unsigned long long ga = (unsigned long long)(size_t)gsrc;
  uv4 g0;
  g0[0] = 1u;                                        // count=1, user descriptor
  g0[1] = lds_addr;                                  // lds_addr [63:32]
  g0[2] = (unsigned)(ga & 0xFFFFFFFFu);              // global_addr lo
  g0[3] = (unsigned)((ga >> 32) & 0x01FFFFFFu) | (2u << 30);  // ga hi | type=2
  unsigned dim0 = 32u;                               // tensor_dim0 (OOB in x)
  unsigned dim1 = (unsigned)rows;                    // tensor_dim1 (OOB in y)
  iv8 g1;
  g1[0] = (int)((1u << 16) |                         // data_size = 2 bytes
                (1u << 20) |                         // pad_enable
                (3u << 22) |                         // pad_interval: 16 DWORDs
                (3u << 25));                         // pad_amount:   4 DWORDs
  g1[1] = (int)((dim0 & 0xFFFFu) << 16);             // dim0[15:0] @ bits[63:48]
  g1[2] = (int)(((dim0 >> 16) & 0xFFFFu) |           // dim0[31:16]
                ((dim1 & 0xFFFFu) << 16));           // dim1[15:0]
  g1[3] = (int)(((dim1 >> 16) & 0xFFFFu) |           // dim1[31:16]
                (32u << 16));                        // tile_dim0 = 32
  g1[4] = (int)128u;                                 // tile_dim1 = 128 rows
  g1[5] = (int)(unsigned)lda;                        // tensor_dim0_stride lo
  g1[6] = 0;                                         // stride hi / dim1_stride
  g1[7] = 0;
  iv4 g2 = {0, 0, 0, 0}, g3 = {0, 0, 0, 0};
#if __clang_major__ >= 23
  iv8 g4 = {0, 0, 0, 0, 0, 0, 0, 0};
  __builtin_amdgcn_tensor_load_to_lds(g0, g1, g2, g3, g4, 0);
#else
  __builtin_amdgcn_tensor_load_to_lds(g0, g1, g2, g3, 0);
#endif
#else
  (void)lds; (void)gsrc; (void)rows; (void)lda;
#endif
}
__device__ __forceinline__ void tdm_wait() {
#if USE_TDM
  __builtin_amdgcn_s_wait_tensorcnt(0);
#endif
}
__device__ __forceinline__ void stage_wait() {
  async_wait();
  tdm_wait();
}

__device__ __forceinline__ u16 f2bf(float f) {
  unsigned u = __float_as_uint(f);
  u += 0x7FFFu + ((u >> 16) & 1u);
  return (u16)(u >> 16);
}
__device__ __forceinline__ float bf2f(u16 b) {
  return __uint_as_float((unsigned)b << 16);
}
__device__ __forceinline__ v8f v8f_zero() {
  v8f z;
  #pragma unroll
  for (int i = 0; i < 8; i++) z[i] = 0.0f;
  return z;
}
__device__ __forceinline__ v8f wmma_bf16(const Frag& a, const Frag& b, v8f c) {
  return __builtin_amdgcn_wmma_f32_16x16x32_bf16(false, a.v, false, b.v,
                                                 (short)0, c, false, false);
}

// --------------------------------------------------------- tile staging
// A tile: 128 rows x 32 k (bf16) -> LDS [row][k], stride 40 (per-lane async)
__device__ __forceinline__ void stage_A(u16* As, const u16* A, int lda,
                                        int m0, int M, int k0, int tid) {
  for (int i = tid; i < 512; i += 256) {
    int r = i >> 2, c = (i & 3) * 8;
    int gr = m0 + r;
    if (gr < M) {
      copy16_to_lds(&As[r * 40 + c], &A[(size_t)gr * lda + k0 + c]);
    } else {
      P16 z = {0u, 0u, 0u, 0u};
      *(P16*)&As[r * 40 + c] = z;
    }
  }
}
// A tile via TDM when available (wave 0 issues the descriptor)
__device__ __forceinline__ void stage_A_auto(u16* As, const u16* A, int lda,
                                             int m0, int M, int k0, int tid,
                                             int wave) {
#if USE_TDM
  (void)tid;
  if (wave == 0)
    tdm_load_A(As, A + (size_t)m0 * lda + k0, M - m0, lda);
#else
  (void)wave;
  stage_A(As, A, lda, m0, M, k0, tid);
#endif
}
// gathered A rows via perm (MoE)
__device__ __forceinline__ void stage_A_gather(u16* As, const u16* X,
                                               const int* perm, int off,
                                               int m0, int cnt, int k0, int tid) {
  for (int i = tid; i < 512; i += 256) {
    int r = i >> 2, c = (i & 3) * 8;
    int gr = m0 + r;
    if (gr < cnt) {
      int tok = perm[off + gr];
      copy16_to_lds(&As[r * 40 + c], &X[(size_t)tok * cD + k0 + c]);
    } else {
      P16 z = {0u, 0u, 0u, 0u};
      *(P16*)&As[r * 40 + c] = z;
    }
  }
}
// B tile: 32 k-rows x 128 n (f32 -> bf16) -> LDS [n][k], stride 40
__device__ __forceinline__ void stage_B(u16* Bs, const float* Bw, int ldb,
                                        int n0, int k0, int tid) {
  for (int i = tid; i < 1024; i += 256) {
    int r = i >> 5, c = (i & 31) * 4;
    F4 v = *(const F4*)&Bw[(size_t)(k0 + r) * ldb + n0 + c];
    Bs[(c + 0) * 40 + r] = f2bf(v.x);
    Bs[(c + 1) * 40 + r] = f2bf(v.y);
    Bs[(c + 2) * 40 + r] = f2bf(v.z);
    Bs[(c + 3) * 40 + r] = f2bf(v.w);
  }
}
// one K-step of WMMA over a 32x64 wave tile (2 m-tiles x 4 n-tiles)
__device__ __forceinline__ void compute_tile(const u16* As, const u16* Bs,
                                             int wm, int wn, int nl, int ka,
                                             int kbv, v8f (&acc)[2][4]) {
  Frag af[2], bf[4];
  #pragma unroll
  for (int i = 0; i < 2; i++) {
    int row = wm + i * 16 + nl;
    af[i].q[0] = *(const P16*)&As[row * 40 + ka];
    af[i].q[1] = *(const P16*)&As[row * 40 + ka + 8];
  }
  #pragma unroll
  for (int j = 0; j < 4; j++) {
    int col = wn + j * 16 + nl;
    bf[j].q[0] = *(const P16*)&Bs[col * 40 + kbv];
    bf[j].q[1] = *(const P16*)&Bs[col * 40 + kbv + 8];
  }
  #pragma unroll
  for (int i = 0; i < 2; i++)
    #pragma unroll
    for (int j = 0; j < 4; j++)
      acc[i][j] = wmma_bf16(af[i], bf[j], acc[i][j]);
}

// ---------------------------------------------------------------- rmsnorm+cast
__global__ __launch_bounds__(256) void rmsnorm_cast_kernel(
    const float* __restrict__ x, const float* __restrict__ w, u16* __restrict__ o)
{
  const int t = blockIdx.x;
  const float* xr = x + (size_t)t * cD;
  float ss = 0.f;
  for (int d = threadIdx.x; d < cD; d += 256) { float v = xr[d]; ss += v * v; }
  __shared__ float red[256];
  red[threadIdx.x] = ss;
  __syncthreads();
  for (int s = 128; s > 0; s >>= 1) {
    if (threadIdx.x < s) red[threadIdx.x] += red[threadIdx.x + s];
    __syncthreads();
  }
  const float rs = rsqrtf(red[0] / (float)cD + 1e-5f);
  for (int d = threadIdx.x; d < cD; d += 256)
    o[(size_t)t * cD + d] = f2bf(xr[d] * rs * w[d]);
}

// ------------------------------------------------- generic bf16 WMMA GEMM
// C[M,N](f32) = A_bf16[M,K] * B_f32->bf16[K,N]  (+ optional residual)
// block tile 128x128, wave tile 32x64, double-buffered LDS, TDM/async A staging
__global__ __launch_bounds__(256) void gemm_bf16_kernel(
    const u16* __restrict__ A, int lda,
    const float* __restrict__ Bw, int ldb,
    float* __restrict__ C, int ldc,
    int M, int N, int K, const float* __restrict__ resid)
{
  __shared__ alignas(16) u16 As[2][128 * 40];
  __shared__ alignas(16) u16 Bs[2][128 * 40];
  const int tid = threadIdx.x, lane = tid & 31, wave = tid >> 5;
  const int wm = (wave & 3) * 32, wn = (wave >> 2) * 64;
  const int m0 = blockIdx.y * 128, n0 = blockIdx.x * 128;
  const int nl = lane & 15;
  const int ka  = (lane < 16) ? 0 : 16;   // A-frag K base
  const int kbv = (lane < 16) ? 0 : 16;   // B-frag K base
  const int rh  = (lane < 16) ? 0 : 8;    // C-row half offset
  v8f acc[2][4];
  for (int i = 0; i < 2; i++) for (int j = 0; j < 4; j++) acc[i][j] = v8f_zero();

  stage_A_auto(As[0], A, lda, m0, M, 0, tid, wave);
  stage_B(Bs[0], Bw, ldb, n0, 0, tid);
  stage_wait();
  __syncthreads();
  int cur = 0;
  for (int k0 = 0; k0 < K; k0 += 32) {
    const int nxt = cur ^ 1;
    if (k0 + 32 < K) {
      stage_A_auto(As[nxt], A, lda, m0, M, k0 + 32, tid, wave);
      stage_B(Bs[nxt], Bw, ldb, n0, k0 + 32, tid);
      __builtin_prefetch(&Bw[(size_t)(k0 + 64) * ldb + n0], 0, 1);
    }
    compute_tile(As[cur], Bs[cur], wm, wn, nl, ka, kbv, acc);
    stage_wait();
    __syncthreads();
    cur = nxt;
  }
  for (int i = 0; i < 2; i++) for (int j = 0; j < 4; j++) {
    int rb = m0 + wm + i * 16 + rh;
    int col = n0 + wn + j * 16 + nl;
    #pragma unroll
    for (int v = 0; v < 8; v++) {
      int row = rb + v;
      if (row < M) {
        float val = acc[i][j][v];
        size_t ix = (size_t)row * ldc + col;
        if (resid) val += resid[ix];
        C[ix] = val;
      }
    }
  }
}

// ---------------------------------------------------------------- RoPE + pack
__global__ __launch_bounds__(256) void rope_pack_kernel(
    const float* __restrict__ qf, const float* __restrict__ kf,
    const float* __restrict__ vf, u16* __restrict__ qb,
    u16* __restrict__ kb, u16* __restrict__ vb)
{
  const int TQ = cT * cH * (cHD / 2);
  const int TK = cT * cKV * (cHD / 2);
  int idx = blockIdx.x * 256 + threadIdx.x;
  const float LNTH = 13.815510557964274f;  // ln(1e6)
  if (idx < TQ) {
    int i = idx & 63, h = (idx >> 6) & (cH - 1);
    int s = (idx >> 10) & (cS - 1), b = idx >> 21;
    float inv = __expf(-((float)(2 * i) / (float)cHD) * LNTH);
    float sn, c; __sincosf((float)s * inv, &sn, &c);
    const float* src = qf + ((size_t)(b * cS + s) * cH + h) * cHD;
    float x1 = src[i], x2 = src[i + 64];
    u16* dst = qb + ((size_t)(b * cH + h) * cS + s) * cHD;
    dst[i] = f2bf(x1 * c - x2 * sn);
    dst[i + 64] = f2bf(x2 * c + x1 * sn);
  } else if (idx < TQ + TK) {
    int j = idx - TQ;
    int i = j & 63, h = (j >> 6) & (cKV - 1);
    int s = (j >> 8) & (cS - 1), b = j >> 19;
    float inv = __expf(-((float)(2 * i) / (float)cHD) * LNTH);
    float sn, c; __sincosf((float)s * inv, &sn, &c);
    const float* src = kf + ((size_t)(b * cS + s) * cKV + h) * cHD;
    float x1 = src[i], x2 = src[i + 64];
    u16* dst = kb + ((size_t)(b * cKV + h) * cS + s) * cHD;
    dst[i] = f2bf(x1 * c - x2 * sn);
    dst[i + 64] = f2bf(x2 * c + x1 * sn);
  } else if (idx < TQ + 2 * TK) {
    int j = idx - TQ - TK;
    int i = j & 63, h = (j >> 6) & (cKV - 1);
    int s = (j >> 8) & (cS - 1), b = j >> 19;
    const float* src = vf + ((size_t)(b * cS + s) * cKV + h) * cHD;
    u16* dst = vb + ((size_t)(b * cKV + h) * cS + s) * cHD;
    dst[i] = f2bf(src[i]);
    dst[i + 64] = f2bf(src[i + 64]);
  }
}

// --------------------------------------------------------- flash attention
__global__ __launch_bounds__(128) void flash_attn_kernel(
    const u16* __restrict__ qb, const u16* __restrict__ kb,
    const u16* __restrict__ vb, u16* __restrict__ ob)
{
  __shared__ alignas(16) u16 Qs[64 * 136];
  __shared__ alignas(16) u16 Ks[64 * 136];
  __shared__ alignas(16) u16 Vs[cHD * 72];   // transposed [hd][key]
  __shared__ alignas(16) u16 Ps[64 * 72];    // per-wave 16-row P tiles
  const int b = blockIdx.z, h = blockIdx.y, qt = blockIdx.x;
  const int kvh = h / (cH / cKV);
  const int tid = threadIdx.x, lane = tid & 31, wave = tid >> 5;
  const int nl = lane & 15;
  const int ka  = (lane < 16) ? 0 : 16;
  const int kbv = (lane < 16) ? 0 : 16;
  const int rh  = (lane < 16) ? 0 : 8;
  const int q0 = qt * 64, mrow = wave * 16;
  const u16* qp = qb + (size_t)(b * cH + h) * cS * cHD;
  const u16* kp = kb + (size_t)(b * cKV + kvh) * cS * cHD;
  const u16* vp = vb + (size_t)(b * cKV + kvh) * cS * cHD;

  for (int i = tid; i < 1024; i += 128) {     // Q tile (async)
    int r = i >> 4, c = (i & 15) * 8;
    copy16_to_lds(&Qs[r * 136 + c], &qp[(size_t)(q0 + r) * cHD + c]);
  }
  float m[8], l[8];
  #pragma unroll
  for (int v = 0; v < 8; v++) { m[v] = -1e30f; l[v] = 0.f; }
  v8f accO[8];
  #pragma unroll
  for (int t = 0; t < 8; t++) accO[t] = v8f_zero();
  const float scale = 0.08838834764831845f;  // 1/sqrt(128)

  for (int kt = 0; kt <= qt; kt++) {
    const int k0 = kt * 64;
    for (int i = tid; i < 1024; i += 128) {   // K tile (async)
      int r = i >> 4, c = (i & 15) * 8;
      copy16_to_lds(&Ks[r * 136 + c], &kp[(size_t)(k0 + r) * cHD + c]);
    }
    for (int i = tid; i < 1024; i += 128) {   // V transposed into LDS
      int r = i >> 4, c = (i & 15) * 8;
      P16 pk = *(const P16*)&vp[(size_t)(k0 + r) * cHD + c];
      const u16* e = (const u16*)&pk;
      #pragma unroll
      for (int j = 0; j < 8; j++) Vs[(c + j) * 72 + r] = e[j];
    }
    async_wait();
    __syncthreads();
    // scores: Q[16x128] x K^T -> 4 n-tiles of 16 keys
    v8f sacc[4];
    #pragma unroll
    for (int t = 0; t < 4; t++) sacc[t] = v8f_zero();
    for (int kk = 0; kk < cHD; kk += 32) {
      Frag a;
      int ar = mrow + nl;
      a.q[0] = *(const P16*)&Qs[ar * 136 + kk + ka];
      a.q[1] = *(const P16*)&Qs[ar * 136 + kk + ka + 8];
      #pragma unroll
      for (int t = 0; t < 4; t++) {
        Frag bf;
        int br = t * 16 + nl;
        bf.q[0] = *(const P16*)&Ks[br * 136 + kk + kbv];
        bf.q[1] = *(const P16*)&Ks[br * 136 + kk + kbv + 8];
        sacc[t] = wmma_bf16(a, bf, sacc[t]);
      }
    }
    // online softmax, per accumulator row (row = mrow + rh + v)
    #pragma unroll
    for (int v = 0; v < 8; v++) {
      const int qrow = q0 + mrow + rh + v;
      float sv[4], rmax = -1e30f;
      #pragma unroll
      for (int t = 0; t < 4; t++) {
        float s = sacc[t][v] * scale;
        if (k0 + t * 16 + nl > qrow) s = -1e30f;   // causal mask
        sv[t] = s;
        rmax = fmaxf(rmax, s);
      }
      for (int o = 1; o < 16; o <<= 1) rmax = fmaxf(rmax, __shfl_xor(rmax, o, 32));
      const float mn = fmaxf(m[v], rmax);
      const float alpha = __expf(m[v] - mn);
      float rsum = 0.f;
      #pragma unroll
      for (int t = 0; t < 4; t++) {
        float p = __expf(sv[t] - mn);
        rsum += p;
        Ps[(mrow + rh + v) * 72 + t * 16 + nl] = f2bf(p);
      }
      for (int o = 1; o < 16; o <<= 1) rsum += __shfl_xor(rsum, o, 32);
      l[v] = l[v] * alpha + rsum;
      m[v] = mn;
      #pragma unroll
      for (int t = 0; t < 8; t++) accO[t][v] = accO[t][v] * alpha;
    }
    // P[16x64] x V[64x128]  (DS ops in-order per wave; Ps is wave-private)
    for (int kk = 0; kk < 64; kk += 32) {
      Frag a;
      int ar = mrow + nl;
      a.q[0] = *(const P16*)&Ps[ar * 72 + kk + ka];
      a.q[1] = *(const P16*)&Ps[ar * 72 + kk + ka + 8];
      #pragma unroll
      for (int t = 0; t < 8; t++) {
        Frag bf;
        int br = t * 16 + nl;
        bf.q[0] = *(const P16*)&Vs[br * 72 + kk + kbv];
        bf.q[1] = *(const P16*)&Vs[br * 72 + kk + kbv + 8];
        accO[t] = wmma_bf16(a, bf, accO[t]);
      }
    }
    __syncthreads();
  }
  for (int t = 0; t < 8; t++) {
    #pragma unroll
    for (int v = 0; v < 8; v++) {
      int srow = q0 + mrow + rh + v;
      int hd = t * 16 + nl;
      float val = accO[t][v] / l[v];
      ob[((size_t)(b * cS + srow) * cH + h) * cHD + hd] = f2bf(val);
    }
  }
}

// ------------------------------------------------------------------ MoE router
__global__ __launch_bounds__(256) void moe_init_kernel(int* counts) {
  if (threadIdx.x < cE) counts[threadIdx.x] = 0;
}

__global__ __launch_bounds__(256) void router_kernel(
    const u16* __restrict__ xn2, const float* __restrict__ gw,
    int* __restrict__ counts, int* __restrict__ aE, float* __restrict__ aW)
{
  const int t = blockIdx.x;
  float part[cE];
  #pragma unroll
  for (int e = 0; e < cE; e++) part[e] = 0.f;
  for (int d = threadIdx.x; d < cD; d += 256) {
    float x = bf2f(xn2[(size_t)t * cD + d]);
    const float* g = gw + (size_t)d * cE;
    #pragma unroll
    for (int e = 0; e < cE; e++) part[e] += x * g[e];
  }
  __shared__ float red[256 * cE];
  #pragma unroll
  for (int e = 0; e < cE; e++) red[threadIdx.x * cE + e] = part[e];
  __syncthreads();
  for (int s = 128; s > 0; s >>= 1) {
    if (threadIdx.x < s)
      #pragma unroll
      for (int e = 0; e < cE; e++)
        red[threadIdx.x * cE + e] += red[(threadIdx.x + s) * cE + e];
    __syncthreads();
  }
  if (threadIdx.x == 0) {
    float lg[cE];
    #pragma unroll
    for (int e = 0; e < cE; e++) lg[e] = red[e];
    int i0 = 0;
    for (int e = 1; e < cE; e++) if (lg[e] > lg[i0]) i0 = e;
    int i1 = (i0 == 0) ? 1 : 0;
    for (int e = 0; e < cE; e++) if (e != i0 && lg[e] > lg[i1]) i1 = e;
    float e0 = 1.0f, e1 = __expf(lg[i1] - lg[i0]);
    float inv = 1.0f / (e0 + e1);
    aE[t * 2] = i0; aE[t * 2 + 1] = i1;
    aW[t * 2] = e0 * inv; aW[t * 2 + 1] = e1 * inv;
    atomicAdd(&counts[i0], 1);
    atomicAdd(&counts[i1], 1);
  }
}

__global__ void moe_scan_kernel(const int* counts, int* offs, int* cursor) {
  if (threadIdx.x == 0 && blockIdx.x == 0) {
    int acc = 0;
    for (int e = 0; e < cE; e++) { offs[e] = acc; cursor[e] = acc; acc += counts[e]; }
  }
}

__global__ __launch_bounds__(256) void moe_scatter_kernel(
    const int* __restrict__ aE, const float* __restrict__ aW,
    int* __restrict__ cursor, int* __restrict__ perm, float* __restrict__ pw)
{
  int t = blockIdx.x * 256 + threadIdx.x;
  if (t < cT) {
    #pragma unroll
    for (int s = 0; s < 2; s++) {
      int e = aE[t * 2 + s];
      int pos = atomicAdd(&cursor[e], 1);
      perm[pos] = t;
      pw[pos] = aW[t * 2 + s];
    }
  }
}

// --------------------------------------- MoE expert GEMM 1: H = silu(XW1)*(XW3)
__global__ __launch_bounds__(256) void moe_gemm1_kernel(
    const u16* __restrict__ X, const float* __restrict__ W1,
    const float* __restrict__ W3, const int* __restrict__ counts,
    const int* __restrict__ offs, const int* __restrict__ perm,
    u16* __restrict__ Hb)
{
  const int e = blockIdx.z;
  const int cnt = counts[e];
  const int m0 = blockIdx.y * 128;
  if (m0 >= cnt) return;
  const int n0 = blockIdx.x * 128;
  const int off = offs[e];
  const float* B1 = W1 + (size_t)e * cD * cF;
  const float* B3 = W3 + (size_t)e * cD * cF;
  __shared__ alignas(16) u16 As[2][128 * 40];
  __shared__ alignas(16) u16 Bs1[2][128 * 40];
  __shared__ alignas(16) u16 Bs3[2][128 * 40];
  const int tid = threadIdx.x, lane = tid & 31, wave = tid >> 5;
  const int wm = (wave & 3) * 32, wn = (wave >> 2) * 64;
  const int nl = lane & 15;
  const int ka = (lane < 16) ? 0 : 16, kbv = (lane < 16) ? 0 : 16;
  const int rh = (lane < 16) ? 0 : 8;
  v8f a1[2][4], a3[2][4];
  for (int i = 0; i < 2; i++) for (int j = 0; j < 4; j++) { a1[i][j] = v8f_zero(); a3[i][j] = v8f_zero(); }

  stage_A_gather(As[0], X, perm, off, m0, cnt, 0, tid);
  stage_B(Bs1[0], B1, cF, n0, 0, tid);
  stage_B(Bs3[0], B3, cF, n0, 0, tid);
  stage_wait();
  __syncthreads();
  int cur = 0;
  for (int k0 = 0; k0 < cD; k0 += 32) {
    const int nxt = cur ^ 1;
    if (k0 + 32 < cD) {
      stage_A_gather(As[nxt], X, perm, off, m0, cnt, k0 + 32, tid);
      stage_B(Bs1[nxt], B1, cF, n0, k0 + 32, tid);
      stage_B(Bs3[nxt], B3, cF, n0, k0 + 32, tid);
    }
    compute_tile(As[cur], Bs1[cur], wm, wn, nl, ka, kbv, a1);
    compute_tile(As[cur], Bs3[cur], wm, wn, nl, ka, kbv, a3);
    stage_wait();
    __syncthreads();
    cur = nxt;
  }
  for (int i = 0; i < 2; i++) for (int j = 0; j < 4; j++) {
    int rb = m0 + wm + i * 16 + rh;
    int col = n0 + wn + j * 16 + nl;
    #pragma unroll
    for (int v = 0; v < 8; v++) {
      int row = rb + v;
      if (row < cnt) {
        float g = a1[i][j][v];
        float hval = g * (1.0f / (1.0f + __expf(-g))) * a3[i][j][v];  // silu*gate
        Hb[(size_t)(off + row) * cF + col] = f2bf(hval);
      }
    }
  }
}

// ------------------------------- MoE expert GEMM 2: out[tok] += w * (H @ W2)
__global__ __launch_bounds__(256) void moe_gemm2_kernel(
    const u16* __restrict__ Hb, const float* __restrict__ W2,
    const int* __restrict__ counts, const int* __restrict__ offs,
    const int* __restrict__ perm, const float* __restrict__ pw,
    float* __restrict__ out)
{
  const int e = blockIdx.z;
  const int cnt = counts[e];
  const int m0 = blockIdx.y * 128;
  if (m0 >= cnt) return;
  const int n0 = blockIdx.x * 128;
  const int off = offs[e];
  const float* Bw = W2 + (size_t)e * cF * cD;
  const u16* Arows = Hb + (size_t)off * cF;
  __shared__ alignas(16) u16 As[2][128 * 40];
  __shared__ alignas(16) u16 Bs[2][128 * 40];
  const int tid = threadIdx.x, lane = tid & 31, wave = tid >> 5;
  const int wm = (wave & 3) * 32, wn = (wave >> 2) * 64;
  const int nl = lane & 15;
  const int ka = (lane < 16) ? 0 : 16, kbv = (lane < 16) ? 0 : 16;
  const int rh = (lane < 16) ? 0 : 8;
  v8f acc[2][4];
  for (int i = 0; i < 2; i++) for (int j = 0; j < 4; j++) acc[i][j] = v8f_zero();

  stage_A_auto(As[0], Arows, cF, m0, cnt, 0, tid, wave);
  stage_B(Bs[0], Bw, cD, n0, 0, tid);
  stage_wait();
  __syncthreads();
  int cur = 0;
  for (int k0 = 0; k0 < cF; k0 += 32) {
    const int nxt = cur ^ 1;
    if (k0 + 32 < cF) {
      stage_A_auto(As[nxt], Arows, cF, m0, cnt, k0 + 32, tid, wave);
      stage_B(Bs[nxt], Bw, cD, n0, k0 + 32, tid);
      __builtin_prefetch(&Bw[(size_t)(k0 + 64) * cD + n0], 0, 1);
    }
    compute_tile(As[cur], Bs[cur], wm, wn, nl, ka, kbv, acc);
    stage_wait();
    __syncthreads();
    cur = nxt;
  }
  for (int i = 0; i < 2; i++) for (int j = 0; j < 4; j++) {
    int rb = m0 + wm + i * 16 + rh;
    int col = n0 + wn + j * 16 + nl;
    #pragma unroll
    for (int v = 0; v < 8; v++) {
      int row = rb + v;
      if (row < cnt) {
        int slot = off + row;
        int tok = perm[slot];
        float wgt = pw[slot];
        atomicAdd(&out[(size_t)tok * cD + col], wgt * acc[i][j][v]);
      }
    }
  }
}

// ------------------------------------------------------------------- launcher
extern "C" void kernel_launch(void* const* d_in, const int* in_sizes, int n_in,
                              void* d_out, int out_size, void* d_ws, size_t ws_size,
                              hipStream_t stream)
{
  (void)in_sizes; (void)n_in; (void)out_size; (void)ws_size;
  const float* hidden = (const float*)d_in[0];
  const float* n1w = (const float*)d_in[1];
  const float* n2w = (const float*)d_in[2];
  const float* wq  = (const float*)d_in[3];
  const float* wk  = (const float*)d_in[4];
  const float* wv  = (const float*)d_in[5];
  const float* wo  = (const float*)d_in[6];
  const float* gw  = (const float*)d_in[7];
  const float* w1  = (const float*)d_in[8];
  const float* w3  = (const float*)d_in[9];
  const float* w2  = (const float*)d_in[10];
  float* out = (float*)d_out;

  char* base = (char*)d_ws;
  size_t off = 0;
  auto carve = [&](size_t bytes) -> void* {
    void* p = base + off;
    off = (off + bytes + 255) & ~(size_t)255;
    return p;
  };
  u16*   xn   = (u16*)  carve((size_t)cT * cD * 2);
  float* qf   = (float*)carve((size_t)cT * cH * cHD * 4);
  float* kf   = (float*)carve((size_t)cT * cKV * cHD * 4);
  float* vf   = (float*)carve((size_t)cT * cKV * cHD * 4);
  u16*   qbp  = (u16*)  carve((size_t)cT * cH * cHD * 2);
  u16*   kbp  = (u16*)  carve((size_t)cT * cKV * cHD * 2);
  u16*   vbp  = (u16*)  carve((size_t)cT * cKV * cHD * 2);
  u16*   ao   = (u16*)  carve((size_t)cT * cH * cHD * 2);
  u16*   xn2  = (u16*)  carve((size_t)cT * cD * 2);
  int*   counts = (int*)carve(cE * 4);
  int*   offs   = (int*)carve(cE * 4);
  int*   cursor = (int*)carve(cE * 4);
  int*   aE   = (int*)  carve((size_t)cT * 2 * 4);
  float* aW   = (float*)carve((size_t)cT * 2 * 4);
  int*   perm = (int*)  carve((size_t)cT * 2 * 4);
  float* pwv  = (float*)carve((size_t)cT * 2 * 4);
  u16*   hbuf = (u16*)  carve((size_t)cT * 2 * cF * 2);

  // 1) norm1 -> bf16
  rmsnorm_cast_kernel<<<cT, 256, 0, stream>>>(hidden, n1w, xn);
  // 2) Q/K/V projections (WMMA)
  gemm_bf16_kernel<<<dim3((cH * cHD) / 128, cT / 128), 256, 0, stream>>>(
      xn, cD, wq, cH * cHD, qf, cH * cHD, cT, cH * cHD, cD, nullptr);
  gemm_bf16_kernel<<<dim3((cKV * cHD) / 128, cT / 128), 256, 0, stream>>>(
      xn, cD, wk, cKV * cHD, kf, cKV * cHD, cT, cKV * cHD, cD, nullptr);
  gemm_bf16_kernel<<<dim3((cKV * cHD) / 128, cT / 128), 256, 0, stream>>>(
      xn, cD, wv, cKV * cHD, vf, cKV * cHD, cT, cKV * cHD, cD, nullptr);
  // 3) RoPE + layout [B,h,S,HD] + bf16 pack
  {
    int total = cT * cH * (cHD / 2) + 2 * cT * cKV * (cHD / 2);
    rope_pack_kernel<<<(total + 255) / 256, 256, 0, stream>>>(qf, kf, vf, qbp, kbp, vbp);
  }
  // 4) flash attention (WMMA both matmuls, online softmax)
  flash_attn_kernel<<<dim3(cS / 64, cH, cB), 128, 0, stream>>>(qbp, kbp, vbp, ao);
  // 5) O projection + residual, writes d_out = hidden + attn@wo
  gemm_bf16_kernel<<<dim3(cD / 128, cT / 128), 256, 0, stream>>>(
      ao, cH * cHD, wo, cD, out, cD, cT, cD, cH * cHD, hidden);
  // 6) norm2 -> bf16
  rmsnorm_cast_kernel<<<cT, 256, 0, stream>>>(out, n2w, xn2);
  // 7) MoE routing: logits -> top2 -> softmax -> gather lists
  moe_init_kernel<<<1, 32, 0, stream>>>(counts);
  router_kernel<<<cT, 256, 0, stream>>>(xn2, gw, counts, aE, aW);
  moe_scan_kernel<<<1, 1, 0, stream>>>(counts, offs, cursor);
  moe_scatter_kernel<<<cT / 256, 256, 0, stream>>>(aE, aW, cursor, perm, pwv);
  // 8) expert SwiGLU (gathered rows, fused w1/w3, WMMA)
  moe_gemm1_kernel<<<dim3(cF / 128, cT / 128, cE), 256, 0, stream>>>(
      xn2, w1, w3, counts, offs, perm, hbuf);
  // 9) expert down-proj, weighted scatter-add into residual output
  moe_gemm2_kernel<<<dim3(cD / 128, cT / 128, cE), 256, 0, stream>>>(
      hbuf, w2, counts, offs, perm, pwv, out);
}
